// SelfAttention_22110491640307
// MI455X (gfx1250) — matrix operand order repaired and checked by reading, not compiled, not run on hardware
//
#include <hip/hip_runtime.h>

typedef __bf16 bf16;
typedef __attribute__((ext_vector_type(16))) __bf16 v16bf;
typedef __attribute__((ext_vector_type(8)))  __bf16 v8bf;
typedef __attribute__((ext_vector_type(8)))  float  v8f;

#define BATCH 8
#define CCH   128
#define NSP   4096
#define HD    64

// ---------- helpers ----------

static __device__ __forceinline__ v8f wmma_bf16(v16bf a, v16bf b, v8f c) {
  // (neg_a, A, neg_b, B, c_mod, C, reuse_a, reuse_b)
  return __builtin_amdgcn_wmma_f32_16x16x32_bf16(false, a, false, b, (short)0, c,
                                                 false, false);
}

// A-operand (16x32 bf16): per lane, 8 contiguous bf16 at p and 8 at p+8.
static __device__ __forceinline__ v16bf load_a_bf16(const bf16* p) {
  v8bf lo = *(const v8bf*)p;
  v8bf hi = *(const v8bf*)(p + 8);
  v16bf r;
#pragma unroll
  for (int k = 0; k < 8; ++k) { r[k] = lo[k]; r[k + 8] = hi[k]; }
  return r;
}

// A-operand built from f32 source (weights): 8 floats at p, 8 at p+8.
static __device__ __forceinline__ v16bf load_a_f32(const float* p) {
  v16bf r;
#pragma unroll
  for (int k = 0; k < 8; ++k) { r[k] = (bf16)p[k]; r[k + 8] = (bf16)p[8 + k]; }
  return r;
}

static __device__ __forceinline__ unsigned pack2(float a, float b) {
  union { bf16 h[2]; unsigned u; } t;
  t.h[0] = (bf16)a; t.h[1] = (bf16)b;
  return t.u;
}

static __device__ __forceinline__ void store_row8(bf16* p, v8f v) {
  v8bf r;
#pragma unroll
  for (int k = 0; k < 8; ++k) r[k] = (bf16)v[k];
  *(v8bf*)p = r;
}

// Build the P^T B-operand (32j x 16i) from two S^T D-tiles: exp, pack to bf16,
// exchange the complementary j-rows with the partner half-wave (xor 16).
static __device__ __forceinline__ v16bf make_p(v8f s0, v8f s1, bool hi) {
  unsigned u0[4], u1[4];
#pragma unroll
  for (int m = 0; m < 4; ++m) {
    u0[m] = pack2(__expf(s0[2 * m]), __expf(s0[2 * m + 1]));
    u1[m] = pack2(__expf(s1[2 * m]), __expf(s1[2 * m + 1]));
  }
  union { unsigned u[8]; v16bf v; } bp;
#pragma unroll
  for (int m = 0; m < 4; ++m) {
    unsigned o0 = (unsigned)__shfl_xor((int)u0[m], 16, 32);
    unsigned o1 = (unsigned)__shfl_xor((int)u1[m], 16, 32);
    bp.u[m]     = hi ? o1 : u0[m];   // j rows 0..7 (lo) / 16..23 (hi)
    bp.u[4 + m] = hi ? u1[m] : o0;   // j rows 8..15 / 24..31
  }
  return bp.v;
}

// ---------- kernel 1: K/Q/H projections (1x1 conv == per-pixel matvec) ----------
// Outputs Ktr/Qtr/Hrow in [b][n][h] layout (h contiguous), bf16.
__global__ void proj_kernel(const float* __restrict__ x,
                            const float* __restrict__ Wk,
                            const float* __restrict__ Wq,
                            const float* __restrict__ Wh,
                            bf16* __restrict__ Ktr, bf16* __restrict__ Qtr,
                            bf16* __restrict__ Hrow) {
  __shared__ bf16 sW[3 * CCH * HD];   // [which][c][h] -> conflict-free reads
  int tid = threadIdx.x;
  for (int idx = tid; idx < CCH * HD; idx += 256) {
    int c = idx >> 6, h = idx & 63;
    sW[idx]                = (bf16)Wk[h * CCH + c];
    sW[CCH * HD + idx]     = (bf16)Wq[h * CCH + c];
    sW[2 * CCH * HD + idx] = (bf16)Wh[h * CCH + c];
  }
  __syncthreads();
  int h = tid & 63, nn = tid >> 6;
  int b = blockIdx.x >> 10;                  // 1024 blocks per batch
  int n = ((blockIdx.x & 1023) << 2) + nn;
  const float* xp = x + (size_t)b * CCH * NSP + n;
  float ak = 0.f, aq = 0.f, ah = 0.f;
#pragma unroll 4
  for (int c = 0; c < CCH; ++c) {
    float xv = xp[c * NSP];                  // broadcast across wave
    ak += (float)sW[c * 64 + h] * xv;
    aq += (float)sW[CCH * HD + c * 64 + h] * xv;
    ah += (float)sW[2 * CCH * HD + c * 64 + h] * xv;
  }
  size_t o = ((size_t)(b * NSP + n)) * HD + h;
  Ktr[o] = (bf16)ak; Qtr[o] = (bf16)aq; Hrow[o] = (bf16)ah;
}

// ---------- kernel 2: Z[j] = sum_i exp(S[i,j]);  S = K^T x Q ----------
// One wave per 32-column block (2 B-tile sets share every A load), loop i.
__global__ void zcol_kernel(const bf16* __restrict__ Ktr,
                            const bf16* __restrict__ Qtr,
                            float* __restrict__ Zb) {
  int lane = threadIdx.x & 31, wave = threadIdx.x >> 5;
  int wid = blockIdx.x * 8 + wave;           // 0..1023
  int b = wid >> 7;
  int j0 = (wid & 127) << 5;                 // 32-wide j tile
  int l15 = lane & 15;
  bool hi = lane >= 16;
  const bf16* Kb = Ktr + (size_t)b * NSP * HD;
  const bf16* Qb = Qtr + (size_t)b * NSP * HD;
  // B tiles: 32(c) x 16(j); lane holds one column, 16 contiguous c
  v16bf Ba0 = *(const v16bf*)(Qb + (size_t)(j0 + l15) * HD + (hi ? 16 : 0));
  v16bf Ba1 = *(const v16bf*)(Qb + (size_t)(j0 + l15) * HD + 32 + (hi ? 16 : 0));
  v16bf Bb0 = *(const v16bf*)(Qb + (size_t)(j0 + 16 + l15) * HD + (hi ? 16 : 0));
  v16bf Bb1 = *(const v16bf*)(Qb + (size_t)(j0 + 16 + l15) * HD + 32 + (hi ? 16 : 0));
  v8f za = {}, zb = {};
  for (int i0 = 0; i0 < NSP; i0 += 16) {
    const bf16* arow = Kb + (size_t)(i0 + l15) * HD + (hi ? 16 : 0);
    v16bf A0 = load_a_bf16(arow);            // c 0..31
    v16bf A1 = load_a_bf16(arow + 32);       // c 32..63
    v8f sa = {}, sb = {};
    sa = wmma_bf16(A0, Ba0, sa); sa = wmma_bf16(A1, Ba1, sa);
    sb = wmma_bf16(A0, Bb0, sb); sb = wmma_bf16(A1, Bb1, sb);
#pragma unroll
    for (int r = 0; r < 8; ++r) {
      za[r] += __expf(sa[r]);
      zb[r] += __expf(sb[r]);
    }
  }
  float ta = za[0] + za[1] + za[2] + za[3] + za[4] + za[5] + za[6] + za[7];
  float tb = zb[0] + zb[1] + zb[2] + zb[3] + zb[4] + zb[5] + zb[6] + zb[7];
  ta += __shfl_xor(ta, 16, 32);              // combine row halves
  tb += __shfl_xor(tb, 16, 32);
  if (lane < 16) {
    Zb[b * NSP + j0 + lane] = ta;
    Zb[b * NSP + j0 + 16 + lane] = tb;
  }
}

// ---------- kernel 2b: Ht[h][n] = Hrow[n][h] / Z[n]  (LDS tile transpose) ----------
__global__ void hdiv_kernel(const bf16* __restrict__ Hrow,
                            const float* __restrict__ Zb,
                            bf16* __restrict__ Htb2) {
  __shared__ bf16 tile[64][68];
  int b = blockIdx.x >> 6;
  int n0 = (blockIdx.x & 63) << 6;
  int tid = threadIdx.x;
#pragma unroll
  for (int k = 0; k < 16; ++k) {
    int idx = tid + 256 * k;
    int nl = idx >> 6, h = idx & 63;
    float v = (float)Hrow[((size_t)(b * NSP + n0 + nl)) * HD + h];
    tile[nl][h] = (bf16)(v / Zb[b * NSP + n0 + nl]);
  }
  __syncthreads();
#pragma unroll
  for (int k = 0; k < 16; ++k) {
    int idx = tid + 256 * k;
    int h = idx >> 6, nl = idx & 63;
    Htb2[((size_t)(b * HD + h)) * NSP + n0 + nl] = tile[nl][h];
  }
}

// ---------- kernel 3: attnT[n][h] = sum_j exp(S[i=n,j]) * Ht[h,j] ----------
// S^T = Q^T x K so the D-tile column (= i) matches the B-operand lane layout
// of the second GEMM; P crosses half-waves with one shfl_xor(16), no LDS.
// 32-wide i tile: every Aq and Ah load feeds two WMMAs (16 WMMA / 16 loads).
__global__ void attn_kernel(const bf16* __restrict__ Ktr,
                            const bf16* __restrict__ Qtr,
                            const bf16* __restrict__ Htb2,
                            bf16* __restrict__ atnT) {
  int lane = threadIdx.x & 31, wave = threadIdx.x >> 5;
  int wid = blockIdx.x * 8 + wave;           // 0..1023
  int b = wid >> 7;
  int i0 = (wid & 127) << 5;                 // 32-wide i tile
  int l15 = lane & 15;
  bool hi = lane >= 16;
  const bf16* Kb = Ktr + (size_t)b * NSP * HD;
  const bf16* Qb = Qtr + (size_t)b * NSP * HD;
  const bf16* Hb = Htb2 + (size_t)b * HD * NSP;
  // B tiles of K (two 16-column sets, fixed for this wave)
  v16bf BKa0 = *(const v16bf*)(Kb + (size_t)(i0 + l15) * HD + (hi ? 16 : 0));
  v16bf BKa1 = *(const v16bf*)(Kb + (size_t)(i0 + l15) * HD + 32 + (hi ? 16 : 0));
  v16bf BKb0 = *(const v16bf*)(Kb + (size_t)(i0 + 16 + l15) * HD + (hi ? 16 : 0));
  v16bf BKb1 = *(const v16bf*)(Kb + (size_t)(i0 + 16 + l15) * HD + 32 + (hi ? 16 : 0));
  v8f accA[4], accB[4];
#pragma unroll
  for (int t = 0; t < 4; ++t) { accA[t] = (v8f){}; accB[t] = (v8f){}; }
  for (int j0 = 0; j0 < NSP; j0 += 32) {
    const bf16* q0 = Qb + (size_t)(j0 + l15) * HD + (hi ? 16 : 0);
    const bf16* q1 = Qb + (size_t)(j0 + 16 + l15) * HD + (hi ? 16 : 0);
    __builtin_prefetch((const void*)(q0 + 32 * HD), 0, 1);
    v16bf Aq00 = load_a_bf16(q0), Aq01 = load_a_bf16(q0 + 32);
    v16bf Aq10 = load_a_bf16(q1), Aq11 = load_a_bf16(q1 + 32);
    v8f sA0 = {}; sA0 = wmma_bf16(Aq00, BKa0, sA0); sA0 = wmma_bf16(Aq01, BKa1, sA0);
    v8f sA1 = {}; sA1 = wmma_bf16(Aq10, BKa0, sA1); sA1 = wmma_bf16(Aq11, BKa1, sA1);
    v8f sB0 = {}; sB0 = wmma_bf16(Aq00, BKb0, sB0); sB0 = wmma_bf16(Aq01, BKb1, sB0);
    v8f sB1 = {}; sB1 = wmma_bf16(Aq10, BKb0, sB1); sB1 = wmma_bf16(Aq11, BKb1, sB1);
    v16bf bpA = make_p(sA0, sA1, hi);        // P^T cols i0..i0+15
    v16bf bpB = make_p(sB0, sB1, hi);        // P^T cols i0+16..i0+31
    // A tiles of Ht: rows h, K-dim j (32 wide), 4 blocks of 16 h; shared by
    // both i half-tiles.
#pragma unroll
    for (int t = 0; t < 4; ++t) {
      const bf16* hp = Hb + (size_t)(16 * t + l15) * NSP + j0 + (hi ? 16 : 0);
      v16bf Ah = load_a_bf16(hp);
      accA[t] = wmma_bf16(Ah, bpA, accA[t]);
      accB[t] = wmma_bf16(Ah, bpB, accB[t]);
    }
  }
  // D tiles: rows h = 16t + r (+8 for hi half), col n = i0 (+16) + l15
  bf16* orowA = atnT + ((size_t)(b * NSP + i0 + l15)) * HD + (hi ? 8 : 0);
  bf16* orowB = atnT + ((size_t)(b * NSP + i0 + 16 + l15)) * HD + (hi ? 8 : 0);
#pragma unroll
  for (int t = 0; t < 4; ++t) {
    store_row8(orowA + 16 * t, accA[t]);
    store_row8(orowB + 16 * t, accB[t]);
  }
}

// ---------- kernel 4: out = scale * (Wv x attn) + x ----------
__global__ void out_kernel(const float* __restrict__ x,
                           const float* __restrict__ Wv,
                           const float* __restrict__ scale,
                           const bf16* __restrict__ atnT,
                           float* __restrict__ out) {
  int lane = threadIdx.x & 31, wave = threadIdx.x >> 5;
  int wid = blockIdx.x * 8 + wave;           // 0..16383
  int b = wid >> 11;
  int rem = wid & 2047;
  int c0 = (rem >> 8) << 4;                  // 8 c-blocks
  int n0 = (rem & 255) << 4;                 // 256 n-blocks
  int l15 = lane & 15;
  bool hi = lane >= 16;
  const float* wrow = Wv + (size_t)(c0 + l15) * HD + (hi ? 16 : 0);
  v16bf A0 = load_a_f32(wrow);               // h 0..31
  v16bf A1 = load_a_f32(wrow + 32);          // h 32..63
  const bf16* arow = atnT + ((size_t)(b * NSP + n0 + l15)) * HD + (hi ? 16 : 0);
  v16bf B0 = *(const v16bf*)(arow);
  v16bf B1 = *(const v16bf*)(arow + 32);
  v8f d = {};
  d = wmma_bf16(A0, B0, d);
  d = wmma_bf16(A1, B1, d);
  float sc = scale[0];
#pragma unroll
  for (int r = 0; r < 8; ++r) {
    int c = c0 + r + (hi ? 8 : 0);
    size_t idx = ((size_t)(b * CCH + c)) * NSP + n0 + l15;
    out[idx] = sc * d[r] + x[idx];
  }
}

// ---------- launch ----------
extern "C" void kernel_launch(void* const* d_in, const int* in_sizes, int n_in,
                              void* d_out, int out_size, void* d_ws, size_t ws_size,
                              hipStream_t stream) {
  (void)in_sizes; (void)n_in; (void)out_size; (void)ws_size;
  const float* x     = (const float*)d_in[0];
  const float* Wk    = (const float*)d_in[1];
  const float* Wq    = (const float*)d_in[2];
  const float* Wh    = (const float*)d_in[3];
  const float* Wv    = (const float*)d_in[4];
  const float* scale = (const float*)d_in[5];

  char* ws = (char*)d_ws;                      // ~20.2 MB total
  bf16*  Ktr  = (bf16*)(ws + 0);               // [B][N][HD] bf16, 4 MB
  bf16*  Qtr  = (bf16*)(ws + 4194304);         // [B][N][HD] bf16, 4 MB
  bf16*  Hrow = (bf16*)(ws + 8388608);         // [B][N][HD] bf16, 4 MB
  bf16*  Htb2 = (bf16*)(ws + 12582912);        // [B][HD][N] bf16, 4 MB
  float* Zb   = (float*)(ws + 16777216);       // [B][N] f32, 128 KB
  bf16*  atnT = (bf16*)(ws + 16908288);        // [B][N][HD] bf16, 4 MB
  float* out  = (float*)d_out;

  proj_kernel<<<BATCH * (NSP / 4), 256, 0, stream>>>(x, Wk, Wq, Wh, Ktr, Qtr, Hrow);
  zcol_kernel<<<BATCH * (NSP / 32) / 8, 256, 0, stream>>>(Ktr, Qtr, Zb);
  hdiv_kernel<<<BATCH * (NSP / 64), 256, 0, stream>>>(Hrow, Zb, Htb2);
  attn_kernel<<<BATCH * (NSP / 32) / 8, 256, 0, stream>>>(Ktr, Qtr, Htb2, atnT);
  out_kernel<<<BATCH * (CCH / 16) * (NSP / 16) / 8, 256, 0, stream>>>(x, Wv, scale, atnT, out);
}
